// DeepAGATConvolution_47974784696359
// MI455X (gfx1250) — matrix operand order repaired and checked by reading, not compiled, not run on hardware
//
#include <hip/hip_runtime.h>
#include <hip/hip_bf16.h>

typedef __attribute__((ext_vector_type(16))) _Float16 v16h;
typedef __attribute__((ext_vector_type(8)))  float    v8f;

#define HDIM   32
#define NHEADS 10
#define LNEG   0.2f
#define EPSV   1e-5f

// ---------------------------------------------------------------------------
// W [64 x 320] fp32 row-major  ->  Wt [320 x 64] f16 (so each WMMA B-fragment
// is 16 contiguous halves = one 32B load per lane)
// ---------------------------------------------------------------------------
__global__ __launch_bounds__(64) void wt_convert(const float* __restrict__ W,
                                                 _Float16* __restrict__ Wt) {
    int idx = blockIdx.x * 64 + threadIdx.x;   // 320*64 = 20480 elements
    if (idx < 320 * 64) {
        int n = idx >> 6, k = idx & 63;
        Wt[n * 64 + k] = (_Float16)W[k * 320 + n];
    }
}

__global__ __launch_bounds__(256) void init_out(const float* __restrict__ bias,
                                                float* __restrict__ out, int total) {
    int i = blockIdx.x * 256 + threadIdx.x;
    if (i < total) out[i] = bias[i & (HDIM - 1)];
}

// Build one f16 A-fragment from a 32-float row: lane holds the contiguous K-run
// [hi*16, hi*16+16), matching the contiguous B fragments from Wt.
__device__ inline v16h cvt_frag(const float* __restrict__ p) {
    const float4 u0 = *(const float4*)(p);
    const float4 u1 = *(const float4*)(p + 4);
    const float4 u2 = *(const float4*)(p + 8);
    const float4 u3 = *(const float4*)(p + 12);
    v16h r;
    r[0]  = (_Float16)u0.x; r[1]  = (_Float16)u0.y; r[2]  = (_Float16)u0.z; r[3]  = (_Float16)u0.w;
    r[4]  = (_Float16)u1.x; r[5]  = (_Float16)u1.y; r[6]  = (_Float16)u1.z; r[7]  = (_Float16)u1.w;
    r[8]  = (_Float16)u2.x; r[9]  = (_Float16)u2.y; r[10] = (_Float16)u2.z; r[11] = (_Float16)u2.w;
    r[12] = (_Float16)u3.x; r[13] = (_Float16)u3.y; r[14] = (_Float16)u3.z; r[15] = (_Float16)u3.w;
    return r;
}

__device__ inline float lrelu(float v) {
    return fmaxf(v, 0.f) + LNEG * fminf(v, 0.f);
}

// ---------------------------------------------------------------------------
// One wave32 per 16-edge tile.
// ---------------------------------------------------------------------------
__global__ __launch_bounds__(32) void agat_edge_kernel(
    const float*    __restrict__ x,      // [N,32]
    const int*      __restrict__ eidx,   // [2,E]  row then col
    const float*    __restrict__ ea,     // [E,32]
    const _Float16* __restrict__ Wt,     // [320,64] f16
    const float*    __restrict__ att,    // [10,64]
    const float*    __restrict__ gamma_, // [10]
    const float*    __restrict__ beta_,  // [10]
    const float*    __restrict__ rmean,  // [10]
    const float*    __restrict__ rvar,   // [10]
    float*          __restrict__ out,    // [N,32]
    int E)
{
    __shared__ float ylds[16 * NHEADS * HDIM];   // leaky(x_j@W) tile, [16][320]
    __shared__ float alds[16 * NHEADS];          // attention logits/weights

    const int L    = threadIdx.x;      // 0..31
    const int m    = L & 15;           // edge-in-tile / N-column
    const int hi   = L >> 4;           // half-wave id
    const int e0   = blockIdx.x * 16;

    // gather indices for this tile (clamped for a possible ragged tail)
    const int eg    = (e0 + m < E) ? (e0 + m) : (E - 1);
    const int ridx  = eidx[eg];
    const int cidx  = eidx[E + eg];

    // A-fragments: kchunk0 = node features, kchunk1 = edge_attr (shared i/j)
    const v16h ai0 = cvt_frag(x  + (size_t)ridx * HDIM + hi * 16);
    const v16h aj0 = cvt_frag(x  + (size_t)cidx * HDIM + hi * 16);
    const v16h ae1 = cvt_frag(ea + (size_t)eg   * HDIM + hi * 16);

    #pragma unroll 1
    for (int h = 0; h < NHEADS; ++h) {
        float p[8];
        #pragma unroll
        for (int v = 0; v < 8; ++v) p[v] = 0.f;

        #pragma unroll
        for (int nt = 0; nt < 2; ++nt) {
            const int col0 = h * HDIM + nt * 16;
            // B fragments: 16 contiguous halves per lane from Wt
            const v16h b0 = *(const v16h*)(Wt + (size_t)(col0 + m) * 64 + hi * 16);
            const v16h b1 = *(const v16h*)(Wt + (size_t)(col0 + m) * 64 + 32 + hi * 16);

            v8f di = {};
            di = __builtin_amdgcn_wmma_f32_16x16x32_f16(false, ai0, false, b0, (short)0, di, false, false);
            di = __builtin_amdgcn_wmma_f32_16x16x32_f16(false, ae1, false, b1, (short)0, di, false, false);
            v8f dj = {};
            dj = __builtin_amdgcn_wmma_f32_16x16x32_f16(false, aj0, false, b0, (short)0, dj, false, false);
            dj = __builtin_amdgcn_wmma_f32_16x16x32_f16(false, ae1, false, b1, (short)0, dj, false, false);

            const float wi = att[h * 64 + nt * 16 + m];          // x_i half of att
            const float wj = att[h * 64 + 32 + nt * 16 + m];     // x_j half of att

            #pragma unroll
            for (int v = 0; v < 8; ++v) {
                // D layout: lanes 0-15 row v, lanes 16-31 row 8+v, col = m
                const float yi = lrelu(di[v]);
                const float yj = lrelu(dj[v]);
                ylds[(hi * 8 + v) * (NHEADS * HDIM) + col0 + m] = yj;
                p[v] += yi * wi + yj * wj;
            }
        }
        // reduce over the 16 columns held by each 16-lane half
        #pragma unroll
        for (int off = 1; off < 16; off <<= 1) {
            #pragma unroll
            for (int v = 0; v < 8; ++v) p[v] += __shfl_xor(p[v], off, 32);
        }
        if (m == 0) {
            #pragma unroll
            for (int v = 0; v < 8; ++v)
                alds[(hi * 8 + v) * NHEADS + h] = p[v];
        }
    }
    __syncthreads();

    // leaky -> per-head norm (GROUPS==HEADS) -> softmax over heads, *0.1 mean
    if (hi == 0) {
        float a[NHEADS];
        float mx = -3.0e38f;
        #pragma unroll
        for (int h = 0; h < NHEADS; ++h) {
            float t = lrelu(alds[m * NHEADS + h]);
            t = (t - rmean[h]) * rsqrtf(rvar[h] + EPSV) * gamma_[h] + beta_[h];
            a[h] = t;
            mx = fmaxf(mx, t);
        }
        float s = 0.f;
        #pragma unroll
        for (int h = 0; h < NHEADS; ++h) { a[h] = __expf(a[h] - mx); s += a[h]; }
        const float inv = 0.1f / s;   // fold head-mean into the weight
        #pragma unroll
        for (int h = 0; h < NHEADS; ++h) alds[m * NHEADS + h] = a[h] * inv;
    }
    __syncthreads();

    // Faithful-TF scatter: message (e,h) -> node row[(h*E+e)/10].
    // ~10 consecutive edges share one q, so pre-reduce before each atomic.
    #pragma unroll 1
    for (int h = 0; h < NHEADS; ++h) {
        const long base = (long)h * E + e0;
        long q  = base / 10;
        int  r  = (int)(base - q * 10);
        int  cnt = 0;
        float acc = 0.f;
        #pragma unroll 1
        for (int e = 0; e < 16; ++e) {
            if (e0 + e >= E) break;
            acc += ylds[e * (NHEADS * HDIM) + h * HDIM + L] * alds[e * NHEADS + h];
            ++cnt;
            if (++r == 10) {
                atomicAdd(&out[(size_t)eidx[(int)q] * HDIM + L], acc);
                acc = 0.f; r = 0; cnt = 0; ++q;
            }
        }
        if (cnt > 0)
            atomicAdd(&out[(size_t)eidx[(int)q] * HDIM + L], acc);
    }
}

// ---------------------------------------------------------------------------
extern "C" void kernel_launch(void* const* d_in, const int* in_sizes, int n_in,
                              void* d_out, int out_size, void* d_ws, size_t ws_size,
                              hipStream_t stream) {
    const float* x     = (const float*)d_in[0];
    const int*   eidx  = (const int*)  d_in[1];
    const float* ea    = (const float*)d_in[2];
    const float* W     = (const float*)d_in[3];
    const float* att   = (const float*)d_in[4];
    const float* bias  = (const float*)d_in[5];
    const float* gam   = (const float*)d_in[6];
    const float* bet   = (const float*)d_in[7];
    const float* rmean = (const float*)d_in[8];
    const float* rvar  = (const float*)d_in[9];
    float* out = (float*)d_out;

    const int N = in_sizes[0] / HDIM;
    const int E = in_sizes[1] / 2;

    _Float16* Wt = (_Float16*)d_ws;   // 320*64*2 = 40 KB

    wt_convert<<<(320 * 64 + 63) / 64, 64, 0, stream>>>(W, Wt);
    init_out<<<(N * HDIM + 255) / 256, 256, 0, stream>>>(bias, out, N * HDIM);

    const int tiles = (E + 15) / 16;
    agat_edge_kernel<<<tiles, 32, 0, stream>>>(x, eidx, ea, Wt, att, gam, bet,
                                               rmean, rvar, out, E);
}
